// LLaDABlock_19224273617624
// MI455X (gfx1250) — matrix operand order repaired and checked by reading, not compiled, not run on hardware
//
#include <hip/hip_runtime.h>
#include <hip/hip_bf16.h>

// ---------------------------------------------------------------------------
// LLaDA block (attention + dense-per-expert MoE) for MI455X / gfx1250.
// bf16 WMMA for all large matmuls, flash-attention (no S^2 tensor),
// one-time fp32->bf16 weight cast+transpose so every GEMM is "NT".
// This round: GEMM LDS staging uses GLOBAL_LOAD_ASYNC_TO_LDS_B128 with a
// double-buffered pipeline tracked by ASYNCcnt (s_wait_asynccnt).
// ---------------------------------------------------------------------------

typedef __bf16 bf16;
typedef __attribute__((ext_vector_type(16))) __bf16 v16bf;
typedef __attribute__((ext_vector_type(8)))  __bf16 v8bf;
typedef __attribute__((ext_vector_type(4)))  __bf16 v4bf;
typedef __attribute__((ext_vector_type(8)))  float  v8f;

#define BDIM 2
#define SDIM 2048
#define DDIM 1024
#define HDIM 16
#define DH   64
#define EDIM 8
#define FDIM 4096
#define MTOK (BDIM * SDIM)   // 4096 tokens

// ---------------------------------------------------------------------------
// WMMA fragment loaders. Layouts per CDNA5 ISA 7.12.2 (wave32):
//  A (16x32 bf16): lane l -> row l%16; l<16 holds K={0..7,16..23}, l>=16 K={8..15,24..31}
//  B (32x16 bf16): lane l -> col l%16; l<16 holds K={0..7,16..23}, l>=16 K={8..15,24..31}
//  C/D (16x16 f32): elem r -> row r + 8*(l>=16), col l%16
// ---------------------------------------------------------------------------
__device__ __forceinline__ v16bf frag_a(const bf16* base, int stride, int row0, int lane) {
    int part = lane >> 4;
    const bf16* p = base + (row0 + (lane & 15)) * stride + part * 8;
    v8bf lo = *(const v8bf*)(p);
    v8bf hi = *(const v8bf*)(p + 16);
    v16bf f;
#pragma unroll
    for (int i = 0; i < 8; i++) { f[i] = lo[i]; f[i + 8] = hi[i]; }
    return f;
}

__device__ __forceinline__ v16bf frag_b(const bf16* base, int stride, int row0, int lane) {
    int part = lane >> 4;
    const bf16* p = base + (row0 + (lane & 15)) * stride + part * 8;
    v8bf lo = *(const v8bf*)(p);
    v8bf hi = *(const v8bf*)(p + 16);
    v16bf f;
#pragma unroll
    for (int i = 0; i < 8; i++) { f[i] = lo[i]; f[i + 8] = hi[i]; }
    return f;
}

#define WMMA_BF16(A_, B_, C_) \
    __builtin_amdgcn_wmma_f32_16x16x32_bf16(false, (A_), false, (B_), (short)0, (C_), false, false)

__device__ __forceinline__ float gelu_tanh(float x) {
    const float c = 0.7978845608028654f; // sqrt(2/pi)
    float x3 = x * x * x;
    return 0.5f * x * (1.0f + tanhf(c * (x + 0.044715f * x3)));
}

// Async global->LDS copy of 16B, tracked by ASYNCcnt (CDNA5 §10 async ops).
// GV mode: per-lane 64-bit global address; LDS byte offset (addr[31:0] of the
// generic shared pointer) in a VGPR.
__device__ __forceinline__ void async_g2l_b128(const bf16* gptr, bf16* lptr) {
    unsigned lds = (unsigned)(uintptr_t)lptr;
    unsigned long long ga = (unsigned long long)(uintptr_t)gptr;
    asm volatile("global_load_async_to_lds_b128 %0, %1, off" :: "v"(lds), "v"(ga) : "memory");
}
__device__ __forceinline__ void wait_async0() {
    asm volatile("s_wait_asynccnt 0x0" ::: "memory");
}

// ---------------------------------------------------------------------------
// One-time fp32 -> bf16 cast with transpose: out[b][c*R + r] = in[b][r*C + c]
// ---------------------------------------------------------------------------
__global__ __launch_bounds__(256) void cast_transpose(const float* __restrict__ in,
                                                      bf16* __restrict__ out,
                                                      int R, int C) {
    size_t base = (size_t)blockIdx.y * R * C;
    int idx = blockIdx.x * 256 + threadIdx.x;
    int r = idx / C, c = idx % C;
    out[base + (size_t)c * R + r] = (bf16)in[base + (size_t)r * C + c];
}

// ---------------------------------------------------------------------------
// RMSNorm (D=1024) + cast to bf16. One block per token row.
// ---------------------------------------------------------------------------
__global__ __launch_bounds__(256) void rmsnorm_cast(const float* __restrict__ x,
                                                    const float* __restrict__ w,
                                                    bf16* __restrict__ out) {
    int row = blockIdx.x;
    int t = threadIdx.x;
    const float4 v = ((const float4*)(x + (size_t)row * DDIM))[t];
    float ss = v.x * v.x + v.y * v.y + v.z * v.z + v.w * v.w;
#pragma unroll
    for (int off = 16; off > 0; off >>= 1) ss += __shfl_xor(ss, off);
    __shared__ float red[8];
    if ((t & 31) == 0) red[t >> 5] = ss;
    __syncthreads();
    float tot = 0.f;
#pragma unroll
    for (int i = 0; i < 8; i++) tot += red[i];
    float scale = rsqrtf(tot * (1.0f / (float)DDIM) + 1e-6f);
    const float4 wv = ((const float4*)w)[t];
    v4bf o;
    o[0] = (bf16)(v.x * scale * wv.x);
    o[1] = (bf16)(v.y * scale * wv.y);
    o[2] = (bf16)(v.z * scale * wv.z);
    o[3] = (bf16)(v.w * scale * wv.w);
    *(v4bf*)(out + (size_t)row * DDIM + t * 4) = o;
}

// ---------------------------------------------------------------------------
// Tiled NT bf16 GEMM: C[M,N] = A[M,K] * Bt[N,K]^T, f32 accumulate.
// Macro tile 128x128, K-step 32, 8 waves, wave tile 64x32 (4x2 WMMA tiles).
// Double-buffered LDS fed by GLOBAL_LOAD_ASYNC_TO_LDS_B128 (ASYNCcnt).
// Epilogues:
//   0: store f32           1: store bf16
//   2: +resid, dual f32 store (x1 buffer and d_out base)
//   3: gelu -> bf16        4: d_out += dispatch[row,e] * acc (RMW, serialized)
// ---------------------------------------------------------------------------
#define LDT 40  // 32 K-elems + 8 pad

__device__ __forceinline__ void stage_async(const bf16* __restrict__ A,
                                            const bf16* __restrict__ Bt,
                                            int Kdim, int m0, int n0, int k0,
                                            int tid, bf16* As, bf16* Bs) {
#pragma unroll
    for (int i = 0; i < 2; i++) {
        int e0 = tid + i * 256;
        int row = e0 >> 2;
        int c8 = (e0 & 3) * 8;
        async_g2l_b128(&A[(size_t)(m0 + row) * Kdim + k0 + c8], &As[row * LDT + c8]);
        async_g2l_b128(&Bt[(size_t)(n0 + row) * Kdim + k0 + c8], &Bs[row * LDT + c8]);
    }
}

template <int MODE>
__global__ __launch_bounds__(256) void gemm_nt(const bf16* __restrict__ A,
                                               const bf16* __restrict__ Bt,
                                               int Ndim, int Kdim,
                                               float* __restrict__ Cf,
                                               bf16* __restrict__ Cb,
                                               const float* __restrict__ resid,
                                               float* __restrict__ out2,
                                               const float* __restrict__ disp,
                                               int eIdx) {
    __shared__ bf16 As[2][128 * LDT];
    __shared__ bf16 Bs[2][128 * LDT];

    const int tid = threadIdx.x;
    const int lane = tid & 31;
    const int w = tid >> 5;
    const int wm = (w >> 2) * 64;
    const int wn = (w & 3) * 32;
    const int m0 = blockIdx.y * 128;
    const int n0 = blockIdx.x * 128;

    v8f acc[4][2];
#pragma unroll
    for (int mt = 0; mt < 4; mt++)
#pragma unroll
        for (int nt = 0; nt < 2; nt++)
#pragma unroll
            for (int r = 0; r < 8; r++) acc[mt][nt][r] = 0.0f;

    // prologue: stage k-step 0 into buffer 0
    stage_async(A, Bt, Kdim, m0, n0, 0, tid, As[0], Bs[0]);
    wait_async0();
    __syncthreads();

    int cur = 0;
    for (int k0 = 0; k0 < Kdim; k0 += 32) {
        int nxt = cur ^ 1;
        if (k0 + 32 < Kdim)
            stage_async(A, Bt, Kdim, m0, n0, k0 + 32, tid, As[nxt], Bs[nxt]);

        v16bf af[4], bfg[2];
#pragma unroll
        for (int mt = 0; mt < 4; mt++) af[mt] = frag_a(As[cur], LDT, wm + mt * 16, lane);
#pragma unroll
        for (int nt = 0; nt < 2; nt++) bfg[nt] = frag_b(Bs[cur], LDT, wn + nt * 16, lane);
#pragma unroll
        for (int mt = 0; mt < 4; mt++)
#pragma unroll
            for (int nt = 0; nt < 2; nt++)
                acc[mt][nt] = WMMA_BF16(af[mt], bfg[nt], acc[mt][nt]);

        // own async loads done; ds_loads already forced complete by WMMA use.
        wait_async0();
        __syncthreads();
        cur = nxt;
    }

#pragma unroll
    for (int mt = 0; mt < 4; mt++)
#pragma unroll
        for (int nt = 0; nt < 2; nt++) {
            int col = n0 + wn + nt * 16 + (lane & 15);
#pragma unroll
            for (int r = 0; r < 8; r++) {
                int rowg = m0 + wm + mt * 16 + r + ((lane >> 4) << 3);
                size_t idx = (size_t)rowg * Ndim + col;
                float val = acc[mt][nt][r];
                if (MODE == 0) {
                    Cf[idx] = val;
                } else if (MODE == 1) {
                    Cb[idx] = (bf16)val;
                } else if (MODE == 2) {
                    float s = val + resid[idx];
                    Cf[idx] = s;
                    out2[idx] = s;
                } else if (MODE == 3) {
                    Cb[idx] = (bf16)gelu_tanh(val);
                } else {
                    Cf[idx] += disp[rowg * EDIM + eIdx] * val;
                }
            }
        }
}

// ---------------------------------------------------------------------------
// Flash attention (non-causal). One block per (64 query rows, b*h).
// 4 waves; wave owns 16 query rows. 64-key tiles: K natural in LDS (B-frag
// needs [key][dh], K-contiguous = natural), V stored transposed [dh][key].
// Scores/PV both via bf16 WMMA; online softmax in f32.
// ---------------------------------------------------------------------------
__global__ __launch_bounds__(128) void flash_attn(const bf16* __restrict__ q,
                                                  const bf16* __restrict__ k,
                                                  const bf16* __restrict__ v,
                                                  bf16* __restrict__ ctx) {
    __shared__ bf16 Ks[64 * 72];
    __shared__ bf16 Vt[64 * 72];
    __shared__ bf16 Ps[4][16 * 72];

    const int qblk = blockIdx.x;          // 0..S/64-1
    const int bh = blockIdx.y;            // 0..B*H-1
    const int b = bh >> 4, h = bh & 15;
    const int w = threadIdx.x >> 5;
    const int lane = threadIdx.x & 31;
    const int tid = threadIdx.x;
    const size_t head = (size_t)b * SDIM * DDIM + (size_t)h * DH;

    // Preload Q fragments (rows fixed for whole kernel), direct from global.
    const int qrow = qblk * 64 + w * 16 + (lane & 15);
    const bf16* Qp = q + head + (size_t)qrow * DDIM;
    v16bf aQ[2];
#pragma unroll
    for (int ks = 0; ks < 2; ks++) {
        int part = lane >> 4;
        const bf16* p = Qp + ks * 32 + part * 8;
        v8bf lo = *(const v8bf*)(p);
        v8bf hi = *(const v8bf*)(p + 16);
#pragma unroll
        for (int i = 0; i < 8; i++) { aQ[ks][i] = lo[i]; aQ[ks][i + 8] = hi[i]; }
    }

    float mrow[8], lrow[8];
    v8f o[4];
#pragma unroll
    for (int r = 0; r < 8; r++) { mrow[r] = -3.0e38f; lrow[r] = 0.0f; }
#pragma unroll
    for (int dt = 0; dt < 4; dt++)
#pragma unroll
        for (int r = 0; r < 8; r++) o[dt][r] = 0.0f;

    for (int j0 = 0; j0 < SDIM; j0 += 64) {
        __syncthreads();
        // Stage K tile (natural) and V tile (transposed) into LDS.
#pragma unroll
        for (int i = 0; i < 4; i++) {
            int e0 = tid + i * 128;
            int row = e0 >> 3;
            int c8 = (e0 & 7) * 8;
            v8bf kv = *(const v8bf*)(k + head + (size_t)(j0 + row) * DDIM + c8);
            *(v8bf*)&Ks[row * 72 + c8] = kv;
            v8bf vv = *(const v8bf*)(v + head + (size_t)(j0 + row) * DDIM + c8);
#pragma unroll
            for (int ii = 0; ii < 8; ii++) Vt[(c8 + ii) * 72 + row] = vv[ii];
        }
        __syncthreads();

        // scores S[16 x 64] = Q * K^T (K-dim = Dh = 64 -> two WMMA k-steps)
        v8f sc[4];
#pragma unroll
        for (int nt = 0; nt < 4; nt++) {
            v8f s;
#pragma unroll
            for (int r = 0; r < 8; r++) s[r] = 0.0f;
            s = WMMA_BF16(aQ[0], frag_b(&Ks[0],  72, nt * 16, lane), s);
            s = WMMA_BF16(aQ[1], frag_b(&Ks[32], 72, nt * 16, lane), s);
            sc[nt] = s;
        }

        // online softmax (per-row reductions across 16-lane half-wave groups)
        const float scale = 0.125f; // 1/sqrt(64)
#pragma unroll
        for (int r = 0; r < 8; r++) {
            float mx = -3.0e38f;
#pragma unroll
            for (int nt = 0; nt < 4; nt++) {
                sc[nt][r] *= scale;
                mx = fmaxf(mx, sc[nt][r]);
            }
#pragma unroll
            for (int off = 1; off < 16; off <<= 1) mx = fmaxf(mx, __shfl_xor(mx, off));
            float mn = fmaxf(mrow[r], mx);
            float alpha = __expf(mrow[r] - mn);
            float rs = 0.0f;
#pragma unroll
            for (int nt = 0; nt < 4; nt++) {
                float p = __expf(sc[nt][r] - mn);
                sc[nt][r] = p;
                rs += p;
            }
#pragma unroll
            for (int off = 1; off < 16; off <<= 1) rs += __shfl_xor(rs, off);
            lrow[r] = lrow[r] * alpha + rs;
#pragma unroll
            for (int dt = 0; dt < 4; dt++) o[dt][r] *= alpha;
            mrow[r] = mn;
        }

        // write P (bf16) into this wave's LDS region in A-matrix source order
#pragma unroll
        for (int nt = 0; nt < 4; nt++)
#pragma unroll
            for (int r = 0; r < 8; r++) {
                int prow = r + ((lane >> 4) << 3);
                Ps[w][prow * 72 + nt * 16 + (lane & 15)] = (bf16)sc[nt][r];
            }
        // (same-wave LDS ops are in-order: no barrier needed before re-read)

        // O += P * V  (K-dim = 64 keys -> two WMMA k-steps)
#pragma unroll
        for (int dt = 0; dt < 4; dt++) {
            o[dt] = WMMA_BF16(frag_a(&Ps[w][0],  72, 0, lane),
                              frag_b(&Vt[0],     72, dt * 16, lane), o[dt]);
            o[dt] = WMMA_BF16(frag_a(&Ps[w][32], 72, 0, lane),
                              frag_b(&Vt[32],    72, dt * 16, lane), o[dt]);
        }
    }

    // normalize and store ctx (bf16, token-major [B*S, D], head column block)
#pragma unroll
    for (int dt = 0; dt < 4; dt++)
#pragma unroll
        for (int r = 0; r < 8; r++) {
            int row = qblk * 64 + w * 16 + r + ((lane >> 4) << 3);
            float val = o[dt][r] / lrow[r];
            ctx[head + (size_t)row * DDIM + dt * 16 + (lane & 15)] = (bf16)val;
        }
}

// ---------------------------------------------------------------------------
// Router: logits = xn2 @ wr (1024x8), softmax, top-2, dispatch weights,
// and atomics into stats[0..7]=token counts, [8..15]=prob sums, [16]=sum z^2.
// One wave per token.
// ---------------------------------------------------------------------------
__global__ __launch_bounds__(256) void router_kernel(const bf16* __restrict__ xn2,
                                                     const float* __restrict__ wr,
                                                     float* __restrict__ disp,
                                                     float* __restrict__ stats) {
    int w = threadIdx.x >> 5;
    int lane = threadIdx.x & 31;
    int tok = blockIdx.x * 8 + w;
    const bf16* xr = xn2 + (size_t)tok * DDIM + lane * 32;

    float acc[EDIM];
#pragma unroll
    for (int e = 0; e < EDIM; e++) acc[e] = 0.0f;
    for (int i = 0; i < 32; i++) {
        float xv = (float)xr[i];
        const float* wrow = wr + (size_t)(lane * 32 + i) * EDIM;
#pragma unroll
        for (int e = 0; e < EDIM; e++) acc[e] += xv * wrow[e];
    }
#pragma unroll
    for (int off = 16; off > 0; off >>= 1)
#pragma unroll
        for (int e = 0; e < EDIM; e++) acc[e] += __shfl_xor(acc[e], off);

    if (lane == 0) {
        float m = acc[0];
#pragma unroll
        for (int e = 1; e < EDIM; e++) m = fmaxf(m, acc[e]);
        float p[EDIM];
        float sum = 0.0f;
#pragma unroll
        for (int e = 0; e < EDIM; e++) { p[e] = __expf(acc[e] - m); sum += p[e]; }
        float inv = 1.0f / sum;
#pragma unroll
        for (int e = 0; e < EDIM; e++) p[e] *= inv;
        float z = m + __logf(sum);

        int i1 = 0;
#pragma unroll
        for (int e = 1; e < EDIM; e++) if (p[e] > p[i1]) i1 = e;
        int i2 = (i1 == 0) ? 1 : 0;
#pragma unroll
        for (int e = 0; e < EDIM; e++) if (e != i1 && p[e] > p[i2]) i2 = e;
        float wsum = p[i1] + p[i2];
        float w1v = p[i1] / wsum;
        float w2v = p[i2] / wsum;
#pragma unroll
        for (int e = 0; e < EDIM; e++)
            disp[(size_t)tok * EDIM + e] = (e == i1) ? w1v : ((e == i2) ? w2v : 0.0f);

        atomicAdd(&stats[i1], 1.0f);
        atomicAdd(&stats[i2], 1.0f);
#pragma unroll
        for (int e = 0; e < EDIM; e++) atomicAdd(&stats[EDIM + e], p[e]);
        atomicAdd(&stats[16], z * z);
    }
}

__global__ void zero_stats(float* stats) {
    if (threadIdx.x < 32) stats[threadIdx.x] = 0.0f;
}

__global__ void finalize_loss(const float* __restrict__ stats, float* __restrict__ out_loss) {
    const float invN = 1.0f / (float)MTOK;
    float aux = 0.0f;
    for (int e = 0; e < EDIM; e++)
        aux += (stats[e] * invN) * (stats[EDIM + e] * invN);
    aux *= (float)EDIM / 2.0f; // E * sum / K
    *out_loss = aux + 0.001f * stats[16] * invN;
}

// ---------------------------------------------------------------------------
// Host launch. Workspace layout (~235 MB, 256B-aligned chunks).
// ---------------------------------------------------------------------------
extern "C" void kernel_launch(void* const* d_in, const int* in_sizes, int n_in,
                              void* d_out, int out_size, void* d_ws, size_t ws_size,
                              hipStream_t stream) {
    const float* x     = (const float*)d_in[0];
    const float* ln1_w = (const float*)d_in[1];
    const float* ln2_w = (const float*)d_in[2];
    const float* wq    = (const float*)d_in[3];
    const float* wk    = (const float*)d_in[4];
    const float* wv    = (const float*)d_in[5];
    const float* wo    = (const float*)d_in[6];
    const float* wr    = (const float*)d_in[7];
    const float* w1    = (const float*)d_in[8];
    const float* w2    = (const float*)d_in[9];
    float* out = (float*)d_out;

    size_t off = 0;
    char* ws = (char*)d_ws;
    auto alloc = [&](size_t bytes) -> void* {
        void* p = ws + off;
        off += (bytes + 255) & ~(size_t)255;
        return p;
    };
    bf16* wqt  = (bf16*)alloc((size_t)DDIM * DDIM * 2);
    bf16* wkt  = (bf16*)alloc((size_t)DDIM * DDIM * 2);
    bf16* wvt  = (bf16*)alloc((size_t)DDIM * DDIM * 2);
    bf16* wot  = (bf16*)alloc((size_t)DDIM * DDIM * 2);
    bf16* w1t  = (bf16*)alloc((size_t)EDIM * DDIM * FDIM * 2); // [E][F][D]
    bf16* w2t  = (bf16*)alloc((size_t)EDIM * FDIM * DDIM * 2); // [E][D][F]
    bf16* xn   = (bf16*)alloc((size_t)MTOK * DDIM * 2);
    bf16* qb   = (bf16*)alloc((size_t)MTOK * DDIM * 2);
    bf16* kb   = (bf16*)alloc((size_t)MTOK * DDIM * 2);
    bf16* vb   = (bf16*)alloc((size_t)MTOK * DDIM * 2);
    bf16* ctxb = (bf16*)alloc((size_t)MTOK * DDIM * 2);
    float* x1  = (float*)alloc((size_t)MTOK * DDIM * 4);
    bf16* xn2  = (bf16*)alloc((size_t)MTOK * DDIM * 2);
    float* disp = (float*)alloc((size_t)MTOK * EDIM * 4);
    bf16* hbuf = (bf16*)alloc((size_t)MTOK * FDIM * 2);
    float* stats = (float*)alloc(32 * 4);

    zero_stats<<<1, 32, 0, stream>>>(stats);

    // one-time bf16 cast + transpose of all GEMM weights
    cast_transpose<<<dim3((DDIM * DDIM) / 256, 1), 256, 0, stream>>>(wq, wqt, DDIM, DDIM);
    cast_transpose<<<dim3((DDIM * DDIM) / 256, 1), 256, 0, stream>>>(wk, wkt, DDIM, DDIM);
    cast_transpose<<<dim3((DDIM * DDIM) / 256, 1), 256, 0, stream>>>(wv, wvt, DDIM, DDIM);
    cast_transpose<<<dim3((DDIM * DDIM) / 256, 1), 256, 0, stream>>>(wo, wot, DDIM, DDIM);
    cast_transpose<<<dim3((DDIM * FDIM) / 256, EDIM), 256, 0, stream>>>(w1, w1t, DDIM, FDIM);
    cast_transpose<<<dim3((FDIM * DDIM) / 256, EDIM), 256, 0, stream>>>(w2, w2t, FDIM, DDIM);

    // attention sublayer
    rmsnorm_cast<<<MTOK, 256, 0, stream>>>(x, ln1_w, xn);
    dim3 gQ(DDIM / 128, MTOK / 128);
    gemm_nt<1><<<gQ, 256, 0, stream>>>(xn, wqt, DDIM, DDIM, nullptr, qb, nullptr, nullptr, nullptr, 0);
    gemm_nt<1><<<gQ, 256, 0, stream>>>(xn, wkt, DDIM, DDIM, nullptr, kb, nullptr, nullptr, nullptr, 0);
    gemm_nt<1><<<gQ, 256, 0, stream>>>(xn, wvt, DDIM, DDIM, nullptr, vb, nullptr, nullptr, nullptr, 0);
    flash_attn<<<dim3(SDIM / 64, BDIM * HDIM), 128, 0, stream>>>(qb, kb, vb, ctxb);
    // x1 = x + ctx@wo ; also seed d_out with the MoE residual base
    gemm_nt<2><<<gQ, 256, 0, stream>>>(ctxb, wot, DDIM, DDIM, x1, nullptr, x, out, nullptr, 0);

    // MoE sublayer
    rmsnorm_cast<<<MTOK, 256, 0, stream>>>(x1, ln2_w, xn2);
    router_kernel<<<MTOK / 8, 256, 0, stream>>>(xn2, wr, disp, stats);

    dim3 gH(FDIM / 128, MTOK / 128);
    dim3 gO(DDIM / 128, MTOK / 128);
    for (int e = 0; e < EDIM; e++) {
        const bf16* w1e = w1t + (size_t)e * FDIM * DDIM; // [F][D]
        const bf16* w2e = w2t + (size_t)e * DDIM * FDIM; // [D][F]
        gemm_nt<3><<<gH, 256, 0, stream>>>(xn2, w1e, FDIM, DDIM, nullptr, hbuf, nullptr, nullptr, nullptr, 0);
        gemm_nt<4><<<gO, 256, 0, stream>>>(hbuf, w2e, DDIM, FDIM, out, nullptr, nullptr, nullptr, disp, e);
    }

    finalize_loss<<<1, 1, 0, stream>>>(stats, out + (size_t)MTOK * DDIM);
}
